// DecoderBlock_33569464385824
// MI455X (gfx1250) — matrix operand order repaired and checked by reading, compiled, not run on hardware
//
#include <hip/hip_runtime.h>

// ---------------------------------------------------------------------------
// Types
// ---------------------------------------------------------------------------
typedef __attribute__((ext_vector_type(16))) __bf16        v16bf;
typedef __attribute__((ext_vector_type(8)))  float         v8f;
typedef __attribute__((ext_vector_type(4)))  unsigned int  u32x4;

union BF16Frag {
  u32x4 u4[2];
  v16bf bf;
};

__device__ __forceinline__ v8f wmma_bf16(v16bf a, v16bf b, v8f c) {
  // D = A(16x32 bf16) * B(32x16 bf16) + C(16x16 f32)
  return __builtin_amdgcn_wmma_f32_16x16x32_bf16(false, a, false, b,
                                                 (short)0, c, false, false);
}

// ---------------------------------------------------------------------------
// Problem constants
// ---------------------------------------------------------------------------
constexpr int BB  = 8;
constexpr int TT  = 1024;
constexpr int DD  = 1024;
constexpr int HH  = 16;
constexpr int HSZ = 64;          // head size
constexpr int MM  = BB * TT;     // 8192 rows

// ---------------------------------------------------------------------------
// fp32 -> bf16 conversion (weights)
// ---------------------------------------------------------------------------
__global__ __launch_bounds__(256) void cvt_kernel(const float* __restrict__ in,
                                                  __bf16* __restrict__ out,
                                                  int n) {
  int i = blockIdx.x * 256 + threadIdx.x;
  if (i < n) out[i] = (__bf16)in[i];
}

// ---------------------------------------------------------------------------
// LayerNorm over D=1024, one row per 256-thread block, bf16 output
// ---------------------------------------------------------------------------
__global__ __launch_bounds__(256) void ln_kernel(const float* __restrict__ x,
                                                 const float* __restrict__ g,
                                                 const float* __restrict__ bta,
                                                 __bf16* __restrict__ out) {
  const int row = blockIdx.x;
  const int tid = threadIdx.x;
  const float* xr = x + (size_t)row * DD;
  float v[4];
  float s = 0.f, s2 = 0.f;
#pragma unroll
  for (int i = 0; i < 4; i++) {
    v[i] = xr[tid + i * 256];
    s += v[i];
    s2 += v[i] * v[i];
  }
#pragma unroll
  for (int d = 16; d >= 1; d >>= 1) {
    s  += __shfl_xor(s,  d, 32);
    s2 += __shfl_xor(s2, d, 32);
  }
  __shared__ float rs[8], rs2[8];
  const int w = tid >> 5, lane = tid & 31;
  if (lane == 0) { rs[w] = s; rs2[w] = s2; }
  __syncthreads();
  s = 0.f; s2 = 0.f;
#pragma unroll
  for (int i = 0; i < 8; i++) { s += rs[i]; s2 += rs2[i]; }
  const float mean = s * (1.0f / 1024.0f);
  const float var  = s2 * (1.0f / 1024.0f) - mean * mean;
  const float inv  = rsqrtf(var + 1e-5f);
#pragma unroll
  for (int i = 0; i < 4; i++) {
    const int c = tid + i * 256;
    out[(size_t)row * DD + c] = (__bf16)((v[i] - mean) * inv * g[c] + bta[c]);
  }
}

// ---------------------------------------------------------------------------
// K transpose: k_bf [B*T, D] (head h at cols h*64..h*64+63)
//   -> Kt [B*H, HS, T]  (row = hs, contiguous over t)
// LDS-tiled so both global sides stay 16B-vectorized and coalesced.
// ---------------------------------------------------------------------------
__global__ __launch_bounds__(256) void ktrans_kernel(
    const __bf16* __restrict__ Kin, __bf16* __restrict__ Kt) {
  __shared__ __align__(16) __bf16 lds_t[64 * 72];  // [hs][t], stride 72 (pad)
  const int bh = blockIdx.y;  // b*H + h
  const int b  = bh >> 4;
  const int h  = bh & (HH - 1);
  const int t0 = blockIdx.x * 64;
  const int tid = threadIdx.x;
  const int tl  = tid >> 2;            // 0..63: t-row on load, hs-row on store
  const int c16 = (tid & 3) * 16;      // 16-element chunk

  // load [t][hs] (vectorized, coalesced), scatter transposed into LDS
  const __bf16* src = Kin + (size_t)(b * TT + t0 + tl) * DD + h * HSZ + c16;
  u32x4 a0 = *(const u32x4*)(src);
  u32x4 a1 = *(const u32x4*)(src + 8);
  __bf16 tmp[16];
  *(u32x4*)(tmp)     = a0;
  *(u32x4*)(tmp + 8) = a1;
#pragma unroll
  for (int e = 0; e < 16; e++) lds_t[(c16 + e) * 72 + tl] = tmp[e];
  __syncthreads();

  // write [hs][t] (vectorized, coalesced)
  const __bf16* p = lds_t + tl * 72 + c16;
  u32x4 o0 = *(const u32x4*)(p);
  u32x4 o1 = *(const u32x4*)(p + 8);
  __bf16* dst = Kt + ((size_t)bh * HSZ + tl) * TT + t0 + c16;
  *(u32x4*)(dst)     = o0;
  *(u32x4*)(dst + 8) = o1;
}

// ---------------------------------------------------------------------------
// bf16 WMMA GEMM: C[M,N] = oscale*(A[M,K] @ B[K,N]) (+bias)(+relu)(+residual)
// Block tile 128x128, Kstep 32, 8 waves (4 in M x 2 in N), 32x64 per wave.
// ---------------------------------------------------------------------------
template <bool OUT_BF16, bool HAS_BIAS, bool RELU, bool HAS_RES>
__global__ __launch_bounds__(256) void gemm_bf16_kernel(
    const __bf16* __restrict__ A, const __bf16* __restrict__ B,
    const float* __restrict__ bias, const float* __restrict__ res,
    void* __restrict__ Cout, int M, int N, int K, float oscale) {
  __shared__ __align__(16) __bf16 lds_a[128 * 40];   // [row][k], stride 40 (pad)
  __shared__ __align__(16) __bf16 lds_b[32 * 136];   // [k][n],   stride 136 (pad)

  const int tid  = threadIdx.x;
  const int lane = tid & 31;
  const int w    = tid >> 5;
  const int wm   = w & 3;          // wave M index (0..3) -> 32 rows each
  const int wn   = w >> 2;         // wave N index (0..1) -> 64 cols each
  const int half = lane >> 4;      // A-fragment half-wave select
  const int mrow = lane & 15;
  const int m0 = blockIdx.y * 128;
  const int n0 = blockIdx.x * 128;

  // cooperative tile-load indices (256 threads)
  const int ar = tid >> 1;             // A row 0..127
  const int ac = (tid & 1) * 16;       // A col 0 or 16
  const int bk = tid >> 3;             // B row 0..31
  const int bn = (tid & 7) * 16;       // B col group

  const __bf16* aptr = A + (size_t)(m0 + ar) * K + ac;
  const __bf16* bptr = B + (size_t)bk * N + n0 + bn;

  v8f acc[2][4] = {};

  u32x4 av0 = *(const u32x4*)(aptr);
  u32x4 av1 = *(const u32x4*)(aptr + 8);
  u32x4 bv0 = *(const u32x4*)(bptr);
  u32x4 bv1 = *(const u32x4*)(bptr + 8);

  for (int k0 = 0; k0 < K; k0 += 32) {
    __syncthreads();  // previous iteration's LDS reads complete
    *(u32x4*)(lds_a + ar * 40 + ac)      = av0;
    *(u32x4*)(lds_a + ar * 40 + ac + 8)  = av1;
    *(u32x4*)(lds_b + bk * 136 + bn)     = bv0;
    *(u32x4*)(lds_b + bk * 136 + bn + 8) = bv1;
    __syncthreads();

    if (k0 + 32 < K) {  // software pipeline: prefetch next tile into registers
      const __bf16* an = aptr + (k0 + 32);
      const __bf16* bp = bptr + (size_t)(k0 + 32) * N;
      av0 = *(const u32x4*)(an);
      av1 = *(const u32x4*)(an + 8);
      bv0 = *(const u32x4*)(bp);
      bv1 = *(const u32x4*)(bp + 8);
    }
    if (k0 + 64 < K) {  // pull tile-after-next toward L2 (global_prefetch_b8)
      __builtin_prefetch(aptr + (k0 + 64), 0, 1);
      __builtin_prefetch(bptr + (size_t)(k0 + 64) * N, 0, 1);
    }

    // A fragments: 16x32 bf16; lane half selects K swizzle per ISA table
    v16bf afrag[2];
#pragma unroll
    for (int mi = 0; mi < 2; mi++) {
      const __bf16* p = lds_a + (wm * 32 + mi * 16 + mrow) * 40 + half * 8;
      BF16Frag f;
      f.u4[0] = *(const u32x4*)(p);        // K = half*8 .. +7
      f.u4[1] = *(const u32x4*)(p + 16);   // K = 16+half*8 .. +7
      afrag[mi] = f.bf;
    }
    // B fragments: 32x16 bf16; lane <-> N column, K in table order per lane
    v16bf bfrag[4];
#pragma unroll
    for (int ni = 0; ni < 4; ni++) {
      const __bf16* p = lds_b + (half * 8) * 136 + wn * 64 + ni * 16 + mrow;
      BF16Frag f;
#pragma unroll
      for (int s = 0; s < 16; s++) f.bf[s] = p[((s & 7) + (s >> 3) * 16) * 136];
      bfrag[ni] = f.bf;
    }
#pragma unroll
    for (int mi = 0; mi < 2; mi++)
#pragma unroll
      for (int ni = 0; ni < 4; ni++)
        acc[mi][ni] = wmma_bf16(afrag[mi], bfrag[ni], acc[mi][ni]);
  }

  // epilogue: C layout = row (r + 8*half), col (lane&15)
#pragma unroll
  for (int mi = 0; mi < 2; mi++) {
#pragma unroll
    for (int ni = 0; ni < 4; ni++) {
      const int col = n0 + wn * 64 + ni * 16 + mrow;
      const float bv = HAS_BIAS ? bias[col] : 0.0f;
#pragma unroll
      for (int r = 0; r < 8; r++) {
        const int row = m0 + wm * 32 + mi * 16 + r + 8 * half;
        float v = acc[mi][ni][r] * oscale + bv;
        if (RELU) v = fmaxf(v, 0.0f);
        if (HAS_RES) v += res[(size_t)row * N + col];
        if (OUT_BF16)
          ((__bf16*)Cout)[(size_t)row * N + col] = (__bf16)v;
        else
          ((float*)Cout)[(size_t)row * N + col] = v;
      }
    }
  }
}

// ---------------------------------------------------------------------------
// Fused causal flash attention. One wave per 16-query tile. Keys in blocks
// of 32. Softmax scale (1/sqrt(D)=1/32, exact power of two) is pre-folded
// into the Q projection. Row sums come from an extra ones-matrix WMMA.
// Only the single diagonal block takes the masked path.
// Q/V/O bf16 in [B, T, H*HS]; K pre-transposed to Kt [B*H, HS, T].
// ---------------------------------------------------------------------------
__global__ __launch_bounds__(128) void attn_kernel(const __bf16* __restrict__ Qg,
                                                   const __bf16* __restrict__ Kt,
                                                   const __bf16* __restrict__ Vg,
                                                   __bf16* __restrict__ Og) {
  __shared__ __align__(16) __bf16 pstage[4 * 16 * 40];  // per-wave P restage
  const int lane = threadIdx.x & 31;
  const int w    = threadIdx.x >> 5;
  const int tile = blockIdx.x * 4 + w;      // 0..8191
  const int q0   = (tile & 63) * 16;        // query tile start within T
  const int bh   = tile >> 6;               // b*H + h
  const int h    = bh & (HH - 1);
  const int b    = bh >> 4;
  const int half = lane >> 4;
  const int mrow = lane & 15;
  const size_t base = (size_t)b * TT * DD + (size_t)h * HSZ;
  __bf16* ps = pstage + w * (16 * 40);

  // Q fragments (16 queries x 64 hs), kept in registers for the whole loop
  v16bf qf[2];
#pragma unroll
  for (int c = 0; c < 2; c++) {
    const __bf16* p = Qg + base + (size_t)(q0 + mrow) * DD + c * 32 + half * 8;
    BF16Frag f;
    f.u4[0] = *(const u32x4*)(p);
    f.u4[1] = *(const u32x4*)(p + 16);
    qf[c] = f.bf;
  }
  // Kt row base for this head: lane <-> contraction row hs
  const __bf16* ktb0 = Kt + ((size_t)bh * HSZ + lane) * TT;        // hs = lane
  const __bf16* ktb1 = Kt + ((size_t)bh * HSZ + 32 + lane) * TT;   // hs = 32+lane

  // all-ones B matrix: row sums via WMMA (l += P @ 1)
  v16bf ones;
#pragma unroll
  for (int e = 0; e < 16; e++) ones[e] = (__bf16)1.0f;

  float rm[8];
#pragma unroll
  for (int r = 0; r < 8; r++) rm[r] = -1e30f;
  v8f acc[4] = {};  // 16 queries x 64 hs output accumulator
  v8f accl   = {};  // row-sum accumulator (every column identical)

  auto process_block = [&](int kbase, bool domask) {
    // S = Q @ K^T : two 16x16 score tiles, contraction HS=64 (2 wmma each)
    v8f st[2] = {};
#pragma unroll
    for (int t = 0; t < 2; t++) {
#pragma unroll
      for (int c = 0; c < 2; c++) {
        const __bf16* kp = Kt + ((size_t)bh * HSZ + c * 32 + half * 8) * TT + kbase + t * 16 + mrow;
        BF16Frag kf;  // lane <-> key column, hs in table order per lane
#pragma unroll
        for (int s = 0; s < 16; s++) kf.bf[s] = kp[(size_t)((s & 7) + (s >> 3) * 16) * TT];
        st[t] = wmma_bf16(qf[c], kf.bf, st[t]);
      }
    }

    // causal mask only on the diagonal block
    float sv[2][8];
#pragma unroll
    for (int t = 0; t < 2; t++) {
      const int key = kbase + t * 16 + mrow;
#pragma unroll
      for (int r = 0; r < 8; r++) {
        float v = st[t][r];
        if (domask) {
          const int qrow = q0 + r + 8 * half;
          v = (key > qrow) ? -1e30f : v;
        }
        sv[t][r] = v;
      }
    }

    // online softmax: row max via shfl_xor within the 16-lane half
    float mnew[8], alpha[8];
#pragma unroll
    for (int r = 0; r < 8; r++) {
      float mx = fmaxf(rm[r], fmaxf(sv[0][r], sv[1][r]));
      mx = fmaxf(mx, __shfl_xor(mx, 1, 32));
      mx = fmaxf(mx, __shfl_xor(mx, 2, 32));
      mx = fmaxf(mx, __shfl_xor(mx, 4, 32));
      mx = fmaxf(mx, __shfl_xor(mx, 8, 32));
      mnew[r]  = mx;
      alpha[r] = __expf(rm[r] - mx);
      rm[r]    = mx;
    }
#pragma unroll
    for (int r = 0; r < 8; r++) {
      const float p0 = __expf(sv[0][r] - mnew[r]);
      const float p1 = __expf(sv[1][r] - mnew[r]);
      const int m = r + 8 * half;
      ps[m * 40 + mrow]      = (__bf16)p0;   // restage C-layout -> A-layout
      ps[m * 40 + 16 + mrow] = (__bf16)p1;
      const float a = alpha[r];
      accl[r] *= a;
#pragma unroll
      for (int ni = 0; ni < 4; ni++) acc[ni][r] *= a;
    }

    // P fragment (16 queries x 32 keys) from per-wave LDS scratch
    BF16Frag pf;
    {
      const __bf16* pp = ps + mrow * 40 + half * 8;
      pf.u4[0] = *(const u32x4*)(pp);
      pf.u4[1] = *(const u32x4*)(pp + 16);
    }
    // acc += P @ V ; V fragments: lane <-> hs column, keys in table order
#pragma unroll
    for (int ni = 0; ni < 4; ni++) {
      const __bf16* vp = Vg + base + (size_t)(kbase + half * 8) * DD + ni * 16 + mrow;
      BF16Frag vf;
#pragma unroll
      for (int s = 0; s < 16; s++) vf.bf[s] = vp[(size_t)((s & 7) + (s >> 3) * 16) * DD];
      acc[ni] = wmma_bf16(pf.bf, vf.bf, acc[ni]);
    }
    // l += P @ ones : row-sum without cross-lane reductions
    accl = wmma_bf16(pf.bf, ones, accl);
  };

  const int nfull = q0 >> 5;  // blocks entirely below the diagonal (unmasked)
  for (int kb = 0; kb < nfull; kb++) process_block(kb * 32, false);
  process_block(nfull * 32, true);  // single diagonal (masked) block

  // O = acc / l, written directly into [B,T,D] bf16 (head-concat is free)
#pragma unroll
  for (int ni = 0; ni < 4; ni++) {
#pragma unroll
    for (int r = 0; r < 8; r++) {
      const float o = acc[ni][r] / accl[r];
      Og[base + (size_t)(q0 + r + 8 * half) * DD + ni * 16 + mrow] = (__bf16)o;
    }
  }
}

// ---------------------------------------------------------------------------
// Orchestration
// ---------------------------------------------------------------------------
extern "C" void kernel_launch(void* const* d_in, const int* in_sizes, int n_in,
                              void* d_out, int out_size, void* d_ws,
                              size_t ws_size, hipStream_t stream) {
  (void)in_sizes; (void)n_in; (void)out_size; (void)ws_size;
  const float* x    = (const float*)d_in[0];
  const float* Wq   = (const float*)d_in[1];
  const float* Wk   = (const float*)d_in[2];
  const float* Wv   = (const float*)d_in[3];
  const float* Wo   = (const float*)d_in[4];
  const float* bo   = (const float*)d_in[5];
  const float* W1   = (const float*)d_in[6];
  const float* b1   = (const float*)d_in[7];
  const float* W2   = (const float*)d_in[8];
  const float* b2   = (const float*)d_in[9];
  const float* ln1g = (const float*)d_in[10];
  const float* ln1b = (const float*)d_in[11];
  const float* ln2g = (const float*)d_in[12];
  const float* ln2b = (const float*)d_in[13];

  char* p = (char*)d_ws;
  auto take = [&](size_t bytes) {
    char* r = p;
    p += (bytes + 255) & ~(size_t)255;
    return r;
  };
  __bf16* wq_bf  = (__bf16*)take((size_t)DD * DD * 2);
  __bf16* wk_bf  = (__bf16*)take((size_t)DD * DD * 2);
  __bf16* wv_bf  = (__bf16*)take((size_t)DD * DD * 2);
  __bf16* wo_bf  = (__bf16*)take((size_t)DD * DD * 2);
  __bf16* w1_bf  = (__bf16*)take((size_t)DD * 4 * DD * 2);
  __bf16* w2_bf  = (__bf16*)take((size_t)4 * DD * DD * 2);
  __bf16* h_bf   = (__bf16*)take((size_t)MM * DD * 2);
  __bf16* q_bf   = (__bf16*)take((size_t)MM * DD * 2);
  __bf16* k_bf   = (__bf16*)take((size_t)MM * DD * 2);
  __bf16* kt_bf  = (__bf16*)take((size_t)MM * DD * 2);
  __bf16* v_bf   = (__bf16*)take((size_t)MM * DD * 2);
  __bf16* att_bf = (__bf16*)take((size_t)MM * DD * 2);
  float*  x2     = (float*)take((size_t)MM * DD * 4);
  __bf16* h2_bf  = (__bf16*)take((size_t)MM * DD * 2);
  __bf16* f1_bf  = (__bf16*)take((size_t)MM * 4 * DD * 2);

  auto cvt = [&](const float* src, __bf16* dst, int n) {
    cvt_kernel<<<(n + 255) / 256, 256, 0, stream>>>(src, dst, n);
  };
  cvt(Wq, wq_bf, DD * DD);
  cvt(Wk, wk_bf, DD * DD);
  cvt(Wv, wv_bf, DD * DD);
  cvt(Wo, wo_bf, DD * DD);
  cvt(W1, w1_bf, DD * 4 * DD);
  cvt(W2, w2_bf, 4 * DD * DD);

  const dim3 gD(DD / 128, MM / 128);       // (8, 64)
  const dim3 g4D(4 * DD / 128, MM / 128);  // (32, 64)

  // h = LN1(x)
  ln_kernel<<<MM, 256, 0, stream>>>(x, ln1g, ln1b, h_bf);
  // Q = (h @ Wq) / 32  (softmax scale folded in, exact: power of two)
  gemm_bf16_kernel<true, false, false, false><<<gD, 256, 0, stream>>>(
      h_bf, wq_bf, nullptr, nullptr, q_bf, MM, DD, DD, 0.03125f);
  // K, V = h @ W{k,v}
  gemm_bf16_kernel<true, false, false, false><<<gD, 256, 0, stream>>>(
      h_bf, wk_bf, nullptr, nullptr, k_bf, MM, DD, DD, 1.0f);
  gemm_bf16_kernel<true, false, false, false><<<gD, 256, 0, stream>>>(
      h_bf, wv_bf, nullptr, nullptr, v_bf, MM, DD, DD, 1.0f);
  // Kt[bh][hs][t] for vectorized QK^T fragments
  ktrans_kernel<<<dim3(TT / 64, BB * HH), 256, 0, stream>>>(k_bf, kt_bf);
  // att = causal softmax(QK^T) V, fused
  attn_kernel<<<(BB * HH * (TT / 16)) / 4, 128, 0, stream>>>(q_bf, kt_bf, v_bf,
                                                             att_bf);
  // x2 = x + att @ Wo + bo
  gemm_bf16_kernel<false, true, false, true><<<gD, 256, 0, stream>>>(
      att_bf, wo_bf, bo, x, x2, MM, DD, DD, 1.0f);
  // h2 = LN2(x2)
  ln_kernel<<<MM, 256, 0, stream>>>(x2, ln2g, ln2b, h2_bf);
  // f1 = relu(h2 @ W1 + b1)
  gemm_bf16_kernel<true, true, true, false><<<g4D, 256, 0, stream>>>(
      h2_bf, w1_bf, b1, nullptr, f1_bf, MM, 4 * DD, DD, 1.0f);
  // out = x2 + f1 @ W2 + b2
  gemm_bf16_kernel<false, true, false, true><<<gD, 256, 0, stream>>>(
      f1_bf, w2_bf, b2, x2, (float*)d_out, MM, DD, 4 * DD, 1.0f);
}
